// GATJK_62689342652841
// MI455X (gfx1250) — matrix operand order repaired and checked by reading, not compiled, not run on hardware
//
#include <hip/hip_runtime.h>
#include <hip/hip_bf16.h>

// ---------------------------------------------------------------------------
// GAT (2 layers, 2 heads x 32) + BN/ELU + JumpingKnowledge(max) + classifier
// for MI455X (gfx1250, wave32, WMMA).
//
// GEMMs: v_wmma_f32_16x16x32_f16 with A/B pre-packed into fragment-major
// layout so each lane loads its v16h fragment as one 32-byte vector load.
// Edge softmax: monotone-uint-encoded float atomicMax + float atomicAdd,
// all node tensors L2-resident (192 MB L2).
// ---------------------------------------------------------------------------

typedef _Float16 f16;
typedef __attribute__((ext_vector_type(16))) _Float16 v16h;
typedef __attribute__((ext_vector_type(8)))  float    v8f;

#define HCC 64      // per-layer channels (2 heads * 32)
#define HIDC 32
#define OUTC 40

// ---- monotone order-preserving float <-> uint encoding for atomicMax ------
__device__ __forceinline__ unsigned fenc(float x) {
    unsigned b = __float_as_uint(x);
    return (b & 0x80000000u) ? ~b : (b | 0x80000000u);
}
__device__ __forceinline__ float fdec(unsigned e) {
    return (e & 0x80000000u) ? __uint_as_float(e ^ 0x80000000u)
                             : __uint_as_float(~e);
}

// ---------------------------------------------------------------------------
// Fragment-major packing indices (CDNA5 ISA 7.12.2, wave32):
//  A 16x32 f16: lane = m + ((kk>>3)&1)*16 ; elem = ((kk>>4)&1)*8 + (kk&7)
//  B 32x16 f16: lane = n + ((kk>>3)&1)*16 ; elem = ((kk>>4)&1)*8 + (kk&7)
// Packed storage: frag[...][lane][elem], 16 halves (32 B) per lane.
// ---------------------------------------------------------------------------
__device__ __forceinline__ size_t a_pack_index(int m, int tile, int k, int ksteps) {
    const int kbi  = k >> 5;
    const int kk   = k & 31;
    const int lane = m + ((kk >> 3) & 1) * 16;
    const int elem = ((kk >> 4) & 1) * 8 + (kk & 7);
    return (((size_t)tile * ksteps + kbi) * 32 + lane) * 16 + elem;
}

// Pack f32 activations [M x K] row-major -> f16 A fragments.
__global__ void pack_a_f32(const float* __restrict__ in, f16* __restrict__ out,
                           int M, int K) {
    int i = blockIdx.x * 256 + threadIdx.x;
    if (i >= M * K) return;
    const int n = i / K, k = i - n * K;
    out[a_pack_index(n & 15, n >> 4, k, K >> 5)] = (f16)in[i];
}

// Pack f32 weights [K x 64] row-major -> f16 B fragments.
__global__ void pack_b_f32(const float* __restrict__ in, f16* __restrict__ out,
                           int K) {
    int i = blockIdx.x * 256 + threadIdx.x;
    if (i >= K * HCC) return;
    const int k = i / HCC, n = i - k * HCC;
    const int kbi = k >> 5, kk = k & 31;
    const int nt = n >> 4;
    const int lane = (n & 15) + ((kk >> 3) & 1) * 16;
    const size_t dst = (((size_t)kbi * 4 + nt) * 32 + lane) * 16 + ((kk >> 4) & 1) * 8 + (kk & 7);
    out[dst] = (f16)in[i];
}

// ---------------------------------------------------------------------------
// C[M x 64] = A[M x K] * B[K x 64]; A/B pre-packed fragment-major.
// One wave per 16-row tile; hot loop = global_load_b128 + v_wmma only.
//  C 16x16 f32 layout: VGPR r -> row = r + (lane>=16)*8, col = lane & 15.
// ---------------------------------------------------------------------------
template <int KSTEPS>
__global__ __launch_bounds__(128) void gemm_wmma_n64(
        const f16* __restrict__ Apack, const f16* __restrict__ Bpack,
        float* __restrict__ C, int M) {
    const int wave = threadIdx.x >> 5;
    const int lane = threadIdx.x & 31;
    const int tile = blockIdx.x * 4 + wave;
    if (tile * 16 >= M) return;

    v8f zero = {0.f,0.f,0.f,0.f,0.f,0.f,0.f,0.f};
    v8f acc[4];
    acc[0] = zero; acc[1] = zero; acc[2] = zero; acc[3] = zero;

    const f16* ap = Apack + ((size_t)tile * KSTEPS * 32 + lane) * 16;
    const f16* bp = Bpack + (size_t)lane * 16;

#pragma unroll
    for (int kbi = 0; kbi < KSTEPS; ++kbi) {
        const v16h a = *(const v16h*)(ap + (size_t)kbi * 32 * 16);
#pragma unroll
        for (int nt = 0; nt < 4; ++nt) {
            const v16h b = *(const v16h*)(bp + ((size_t)kbi * 4 + nt) * 32 * 16);
            acc[nt] = __builtin_amdgcn_wmma_f32_16x16x32_f16(
                false, a, false, b, (short)0, acc[nt], false, false);
        }
    }

    const int halfid = lane >> 4;
    const int nm     = lane & 15;
#pragma unroll
    for (int nt = 0; nt < 4; ++nt) {
#pragma unroll
        for (int r = 0; r < 8; ++r) {
            const int orow = tile * 16 + r + halfid * 8;
            const int ocol = nt * 16 + nm;
            C[(size_t)orow * HCC + ocol] = acc[nt][r];
        }
    }
}

// ---------------------------------------------------------------------------
// Per-node attention logits: alpha_src[n,h] = <h[n,h,:], a_s[h,:]> (and dst)
// ---------------------------------------------------------------------------
__global__ void node_alpha(const float* __restrict__ h,
                           const float* __restrict__ a_s, const float* __restrict__ a_d,
                           float* __restrict__ asrc, float* __restrict__ adst, int Nn) {
    int t = blockIdx.x * 256 + threadIdx.x;
    if (t >= Nn * 2) return;
    const int n = t >> 1, hd = t & 1;
    const float* hp = h + (size_t)n * HCC + hd * HIDC;
    float ss = 0.f, sd = 0.f;
#pragma unroll
    for (int i = 0; i < HIDC; ++i) {
        const float v = hp[i];
        ss += v * a_s[hd * HIDC + i];
        sd += v * a_d[hd * HIDC + i];
    }
    asrc[t] = ss;
    adst[t] = sd;
}

// ---------------------------------------------------------------------------
// Edge pass 1: e = leaky_relu(asrc[src]+adst[dst]); segment max over dst.
// ---------------------------------------------------------------------------
__global__ void edge_max(const int* __restrict__ ei, const float* __restrict__ asrc,
                         const float* __restrict__ adst, float* __restrict__ evals,
                         unsigned* __restrict__ menc, int Ee, int Et) {
    int e = blockIdx.x * 256 + threadIdx.x;
    if (e >= Et) return;
    int s, d;
    if (e < Ee) { s = ei[e]; d = ei[Ee + e]; } else { s = d = e - Ee; }
#pragma unroll
    for (int hd = 0; hd < 2; ++hd) {
        float v = asrc[s * 2 + hd] + adst[d * 2 + hd];
        v = v > 0.f ? v : 0.2f * v;
        evals[(size_t)e * 2 + hd] = v;
        atomicMax(&menc[d * 2 + hd], fenc(v));
    }
}

// ---------------------------------------------------------------------------
// Edge pass 2: ee = exp(e - m[dst]) (in place); denom[dst] += ee.
// ---------------------------------------------------------------------------
__global__ void edge_exp(const int* __restrict__ ei, float* __restrict__ evals,
                         const unsigned* __restrict__ menc, float* __restrict__ denom,
                         int Ee, int Et) {
    int e = blockIdx.x * 256 + threadIdx.x;
    if (e >= Et) return;
    int d;
    if (e < Ee) { d = ei[Ee + e]; } else { d = e - Ee; }
#pragma unroll
    for (int hd = 0; hd < 2; ++hd) {
        const float v  = evals[(size_t)e * 2 + hd];
        const float mm = fdec(menc[d * 2 + hd]);
        const float x  = expf(v - mm);
        evals[(size_t)e * 2 + hd] = x;
        atomicAdd(&denom[d * 2 + hd], x);
    }
}

// ---------------------------------------------------------------------------
// Edge pass 3: agg[dst] += h[src] * ee/denom[dst].  16 lanes/edge, 4 ch/lane.
// ---------------------------------------------------------------------------
__global__ void edge_aggr(const int* __restrict__ ei, const float* __restrict__ h,
                          const float* __restrict__ evals, const float* __restrict__ denom,
                          float* __restrict__ agg, int Ee, int Et) {
    const int tid = blockIdx.x * 256 + threadIdx.x;
    const int e   = tid >> 4;
    const int sub = tid & 15;
    if (e >= Et) return;
    int s, d;
    if (e < Ee) { s = ei[e]; d = ei[Ee + e]; } else { s = d = e - Ee; }
    const int c0 = sub * 4;
    const int hd = (c0 >= HIDC) ? 1 : 0;
    const float w = evals[(size_t)e * 2 + hd] / (denom[d * 2 + hd] + 1e-16f);
    const float4 hv = *(const float4*)(h + (size_t)s * HCC + c0);
    float* op = agg + (size_t)d * HCC + c0;
    atomicAdd(op + 0, hv.x * w);
    atomicAdd(op + 1, hv.y * w);
    atomicAdd(op + 2, hv.z * w);
    atomicAdd(op + 3, hv.w * w);
}

// ---------------------------------------------------------------------------
// Layer-1 epilogue: x1 = elu(BN(agg + b1)); emit f32 (for JK) and the f16
// copy directly in A-fragment order for the layer-2 WMMA GEMM (K=64).
// ---------------------------------------------------------------------------
__global__ void bn_elu(const float* __restrict__ agg, const float* __restrict__ b1,
                       const float* __restrict__ gma, const float* __restrict__ bta,
                       const float* __restrict__ mean, const float* __restrict__ var,
                       float* __restrict__ x1f, f16* __restrict__ x1pack, int total) {
    int i = blockIdx.x * 256 + threadIdx.x;
    if (i >= total) return;
    const int c = i & (HCC - 1);
    const int n = i >> 6;
    float v = agg[i] + b1[c];
    v = (v - mean[c]) * rsqrtf(var[c] + 1e-5f) * gma[c] + bta[c];
    v = v > 0.f ? v : expm1f(v);
    x1f[i] = v;
    x1pack[a_pack_index(n & 15, n >> 4, c, 2)] = (f16)v;
}

// ---------------------------------------------------------------------------
// Final: jk = max(x1, agg2 + b2); logits = jk@Wf + bf; log_softmax.
// ---------------------------------------------------------------------------
__global__ __launch_bounds__(128) void final_head(
        const float* __restrict__ x1f, const float* __restrict__ agg2,
        const float* __restrict__ b2, const float* __restrict__ Wf,
        const float* __restrict__ bf, float* __restrict__ out, int Nn) {
    int n = blockIdx.x * 128 + threadIdx.x;
    if (n >= Nn) return;
    float lg[OUTC];
#pragma unroll
    for (int o = 0; o < OUTC; ++o) lg[o] = bf[o];
    for (int c = 0; c < HCC; ++c) {
        const float x2 = agg2[(size_t)n * HCC + c] + b2[c];
        const float jk = fmaxf(x1f[(size_t)n * HCC + c], x2);
#pragma unroll
        for (int o = 0; o < OUTC; ++o) lg[o] += jk * Wf[c * OUTC + o];
    }
    float mx = lg[0];
#pragma unroll
    for (int o = 1; o < OUTC; ++o) mx = fmaxf(mx, lg[o]);
    float sum = 0.f;
#pragma unroll
    for (int o = 0; o < OUTC; ++o) sum += expf(lg[o] - mx);
    const float lse = mx + logf(sum);
#pragma unroll
    for (int o = 0; o < OUTC; ++o) out[(size_t)n * OUTC + o] = lg[o] - lse;
}

// ---------------------------------------------------------------------------
extern "C" void kernel_launch(void* const* d_in, const int* in_sizes, int n_in,
                              void* d_out, int out_size, void* d_ws, size_t ws_size,
                              hipStream_t stream) {
    const float* x    = (const float*)d_in[0];
    const int*   ei   = (const int*)  d_in[1];
    const float* W1   = (const float*)d_in[2];
    const float* as1  = (const float*)d_in[3];
    const float* ad1  = (const float*)d_in[4];
    const float* b1   = (const float*)d_in[5];
    const float* gma  = (const float*)d_in[6];
    const float* bta  = (const float*)d_in[7];
    const float* mean = (const float*)d_in[8];
    const float* var  = (const float*)d_in[9];
    const float* W2   = (const float*)d_in[10];
    const float* as2  = (const float*)d_in[11];
    const float* ad2  = (const float*)d_in[12];
    const float* b2   = (const float*)d_in[13];
    const float* Wf   = (const float*)d_in[14];
    const float* bf   = (const float*)d_in[15];
    float* out = (float*)d_out;

    const int IN = 128;
    const int Nn = in_sizes[0] / IN;      // 50000
    const int Ee = in_sizes[1] / 2;       // 1200000
    const int Et = Ee + Nn;               // + self loops

    // ---- workspace carve-out ---------------------------------------------
    char* wsb = (char*)d_ws;
    size_t off = 0;
    auto take = [&](size_t bytes) -> char* {
        char* p = wsb + off;
        off = (off + bytes + 255) & ~(size_t)255;
        return p;
    };
    float*    h_buf  = (float*)   take((size_t)Nn * HCC * 4);  // per-layer h
    float*    agg    = (float*)   take((size_t)Nn * HCC * 4);  // per-layer aggregation
    float*    x1f    = (float*)   take((size_t)Nn * HCC * 4);  // post BN/ELU f32
    f16*      apack1 = (f16*)     take((size_t)Nn * IN  * 2);  // x in A-frag order
    f16*      apack2 = (f16*)     take((size_t)Nn * HCC * 2);  // x1 in A-frag order
    float*    evals  = (float*)   take((size_t)Et * 2 * 4);    // e / ee (in place)
    float*    asrc   = (float*)   take((size_t)Nn * 2 * 4);
    float*    adst   = (float*)   take((size_t)Nn * 2 * 4);
    unsigned* menc   = (unsigned*)take((size_t)Nn * 2 * 4);
    float*    denom  = (float*)   take((size_t)Nn * 2 * 4);
    f16*      bpack1 = (f16*)     take((size_t)IN  * HCC * 2); // W1 in B-frag order
    f16*      bpack2 = (f16*)     take((size_t)HCC * HCC * 2); // W2 in B-frag order
    (void)ws_size; (void)n_in; (void)out_size;

    const int tiles = (Nn + 15) / 16;
    const dim3 gGemm((tiles + 3) / 4), bGemm(128);
    const dim3 gAlpha((Nn * 2 + 255) / 256), b256(256);
    const dim3 gEdge((Et + 255) / 256);
    const dim3 gAggr(((size_t)Et * 16 + 255) / 256);
    const dim3 gBN((Nn * HCC + 255) / 256);
    const dim3 gFin((Nn + 127) / 128), bFin(128);

    // ---- operand packing (f32 -> f16, fragment-major) --------------------
    pack_b_f32<<<(IN * HCC + 255) / 256, 256, 0, stream>>>(W1, bpack1, IN);
    pack_b_f32<<<(HCC * HCC + 255) / 256, 256, 0, stream>>>(W2, bpack2, HCC);
    pack_a_f32<<<((size_t)Nn * IN + 255) / 256, 256, 0, stream>>>(x, apack1, Nn, IN);

    // =========================== layer 1 ==================================
    hipMemsetAsync(agg,   0, (size_t)Nn * HCC * 4, stream);
    hipMemsetAsync(menc,  0, (size_t)Nn * 2 * 4, stream);
    hipMemsetAsync(denom, 0, (size_t)Nn * 2 * 4, stream);

    gemm_wmma_n64<4><<<gGemm, bGemm, 0, stream>>>(apack1, bpack1, h_buf, Nn);
    node_alpha<<<gAlpha, b256, 0, stream>>>(h_buf, as1, ad1, asrc, adst, Nn);
    edge_max <<<gEdge, b256, 0, stream>>>(ei, asrc, adst, evals, menc, Ee, Et);
    edge_exp <<<gEdge, b256, 0, stream>>>(ei, evals, menc, denom, Ee, Et);
    edge_aggr<<<gAggr, b256, 0, stream>>>(ei, h_buf, evals, denom, agg, Ee, Et);
    bn_elu   <<<gBN, b256, 0, stream>>>(agg, b1, gma, bta, mean, var, x1f, apack2, Nn * HCC);

    // =========================== layer 2 ==================================
    hipMemsetAsync(agg,   0, (size_t)Nn * HCC * 4, stream);
    hipMemsetAsync(menc,  0, (size_t)Nn * 2 * 4, stream);
    hipMemsetAsync(denom, 0, (size_t)Nn * 2 * 4, stream);

    gemm_wmma_n64<2><<<gGemm, bGemm, 0, stream>>>(apack2, bpack2, h_buf, Nn);
    node_alpha<<<gAlpha, b256, 0, stream>>>(h_buf, as2, ad2, asrc, adst, Nn);
    edge_max <<<gEdge, b256, 0, stream>>>(ei, asrc, adst, evals, menc, Ee, Et);
    edge_exp <<<gEdge, b256, 0, stream>>>(ei, evals, menc, denom, Ee, Et);
    edge_aggr<<<gAggr, b256, 0, stream>>>(ei, h_buf, evals, denom, agg, Ee, Et);

    // =================== JK-max + classifier + log_softmax ================
    final_head<<<gFin, bFin, 0, stream>>>(x1f, agg, b2, Wf, bf, out, Nn);
}
